// BiLSTM_CRF_87351044866064
// MI455X (gfx1250) — matrix operand order repaired and checked by reading, not compiled, not run on hardware
//
#include <hip/hip_runtime.h>

typedef __attribute__((ext_vector_type(16))) _Float16 v16h;
typedef __attribute__((ext_vector_type(8)))  float    v8f;
typedef __attribute__((ext_vector_type(4)))  float    v4f;

#define TAGS       32
#define SEQ        2048
#define BATCH      512
#define MAXCHUNKS  16
#define START_TAG  30
#define STOP_TAG   31
#define CSTRIDE    36   // padded LDS row stride (floats) for monolithic kernel
#define HSTRIDE    17   // padded LDS row stride (uints = 2 halfs) for chunk kernel

// Raw-hardware transcendentals: v_exp_f32 / v_log_f32 without denorm fixup code.
__device__ __forceinline__ float nexp(float x) {
    return __builtin_amdgcn_exp2f(x * 1.44269504088896340736f);
}
__device__ __forceinline__ float nlog(float x) {
    return __builtin_amdgcn_logf(x) * 0.69314718055994530942f;
}
__device__ __forceinline__ unsigned int packh2(float a, float b) {
    unsigned short ha = __builtin_bit_cast(unsigned short, (_Float16)a);
    unsigned short hb = __builtin_bit_cast(unsigned short, (_Float16)b);
    return (unsigned int)ha | ((unsigned int)hb << 16);
}

union B32x8 { v16h h; unsigned int u[8]; uint4 v4[2]; };

// ---------------------------------------------------------------------------
// Chunk kernel: one wave per (chunk c = blockIdx.x, batch b = blockIdx.y).
// Computes the 32x32 log-semiring product matrix of `clen` steps in exp space:
//   P <- diag(exp(feat[t])) * Etrans * P, renormalized by tile max each step
// (running log-offset "off"). P is carried as two f16 WMMA B-fragments and
// round-tripped through LDS each step to restore B layout after the D-layout
// WMMA output. 4 x v_wmma_f32_16x16x32_f16 per step (M=N=K=32).
// ---------------------------------------------------------------------------
__global__ __launch_bounds__(32)
void crf_chunk_kernel(const float* __restrict__ feats,
                      const float* __restrict__ trans,
                      unsigned int* __restrict__ chunkBuf,
                      float* __restrict__ offBuf,
                      int clen)
{
    __shared__ __align__(16) unsigned int hbuf[32 * HSTRIDE]; // packed f16 [N=k][i]

    const int  lane = threadIdx.x;
    const bool lo   = lane < 16;
    const int  col  = lane & 15;
    const int  Mb   = lo ? 0 : 8;       // D layout: reg r -> M = Mb + r
    const int  Mh   = Mb >> 1;          // uint offset for packed stores
    const int  Kh   = lo ? 0 : 8;       // B layout uint offset: K = 2*(Kh+q) (+1)
    const int  c    = blockIdx.x;
    const int  b    = blockIdx.y;
    const int  nchunks = gridDim.x;
    const float* featRow = feats + ((size_t)b * SEQ + (size_t)c * clen) * TAGS;

    // Persistent A fragments: A[i][j] = exp(trans[i][j]); elem e -> K=e+(lo?0:16)
    v16h a0, a1;
#pragma unroll
    for (int e = 0; e < 16; ++e) {
        const int j = e + (lo ? 0 : 16);
        a0[e] = (_Float16)nexp(trans[(col)      * TAGS + j]);
        a1[e] = (_Float16)nexp(trans[(col + 16) * TAGS + j]);
    }
    const v8f zc = {};

    // P = Identity in B layout: elem e -> (K = e + (lo?0:16), N = col (+16 tile1))
    B32x8 fb0, fb1;
#pragma unroll
    for (int q = 0; q < 8; ++q) { fb0.u[q] = 0u; fb1.u[q] = 0u; }
    {
        const unsigned int one = 0x3c00u; // f16 1.0
        if (lo) fb0.u[col >> 1] = (col & 1) ? (one << 16) : one;  // K = col == N
        else    fb1.u[col >> 1] = (col & 1) ? (one << 16) : one;  // K = 16+col == N
    }
    float off = 0.0f;

    for (int t = 0; t < clen; ++t) {
        const float* fr = featRow + (size_t)t * TAGS;
        __builtin_prefetch(fr + 16 * TAGS, 0, 1);   // global_prefetch_b8, 2KB ahead

        v8f d00 = __builtin_amdgcn_wmma_f32_16x16x32_f16(false, a0, false, fb0.h, (short)0, zc, false, false);
        v8f d01 = __builtin_amdgcn_wmma_f32_16x16x32_f16(false, a0, false, fb1.h, (short)0, zc, false, false);
        v8f d10 = __builtin_amdgcn_wmma_f32_16x16x32_f16(false, a1, false, fb0.h, (short)0, zc, false, false);
        v8f d11 = __builtin_amdgcn_wmma_f32_16x16x32_f16(false, a1, false, fb1.h, (short)0, zc, false, false);

        // scale rows by exp(feat[t][i])
        float f[32];
#pragma unroll
        for (int q = 0; q < 8; ++q) {
            v4f f4 = *(const v4f*)(fr + q * 4);
            f[q*4+0] = f4.x; f[q*4+1] = f4.y; f[q*4+2] = f4.z; f[q*4+3] = f4.w;
        }
#pragma unroll
        for (int r = 0; r < 8; ++r) {
            const float e0 = nexp(lo ? f[r]      : f[r + 8]);    // i = Mb + r
            const float e1 = nexp(lo ? f[16 + r] : f[24 + r]);   // i = 16 + Mb + r
            d00[r] *= e0;  d01[r] *= e0;
            d10[r] *= e1;  d11[r] *= e1;
        }
        // tile max -> renormalize so values fit f16; accumulate log offset
        float m = d00[0];
#pragma unroll
        for (int r = 0; r < 8; ++r) {
            m = fmaxf(m, d00[r]); m = fmaxf(m, d01[r]);
            m = fmaxf(m, d10[r]); m = fmaxf(m, d11[r]);
        }
#pragma unroll
        for (int s = 16; s > 0; s >>= 1) m = fmaxf(m, __shfl_xor(m, s, 32));
        const float inv = __builtin_amdgcn_rcpf(m);
        off += nlog(m);

        // store packed f16 into hbuf[N][i] (pairs along i), reload in B layout
#pragma unroll
        for (int q = 0; q < 4; ++q) {
            hbuf[(col)      * HSTRIDE + Mh + q]     = packh2(d00[2*q] * inv, d00[2*q+1] * inv);
            hbuf[(col + 16) * HSTRIDE + Mh + q]     = packh2(d01[2*q] * inv, d01[2*q+1] * inv);
            hbuf[(col)      * HSTRIDE + 8 + Mh + q] = packh2(d10[2*q] * inv, d10[2*q+1] * inv);
            hbuf[(col + 16) * HSTRIDE + 8 + Mh + q] = packh2(d11[2*q] * inv, d11[2*q+1] * inv);
        }
#pragma unroll
        for (int q = 0; q < 8; ++q) {
            fb0.u[q] = hbuf[(col)      * HSTRIDE + Kh + q];  // B elem K = 2*(Kh+q)(+1), N = col
            fb1.u[q] = hbuf[(col + 16) * HSTRIDE + Kh + q];  // N = col + 16
        }
    }

    // emit chunk matrix (f16, B-fragment order) + scalar log offset
    unsigned int* outp = chunkBuf + (size_t)(b * nchunks + c) * 512;
    *(uint4*)(outp + lane * 8)           = fb0.v4[0];
    *(uint4*)(outp + lane * 8 + 4)       = fb0.v4[1];
    *(uint4*)(outp + 256 + lane * 8)     = fb1.v4[0];
    *(uint4*)(outp + 256 + lane * 8 + 4) = fb1.v4[1];
    if (lane == 0) offBuf[b * nchunks + c] = off;
}

// ---------------------------------------------------------------------------
// Combine kernel: one wave per batch, lane = tag. Folds chunk matrices into
// the init vector with shuffle-based logsumexp matvecs, applies STOP row.
// Stored element (K=j, N=k) of chunk (b,c) holds expP[j,k] (j = end tag).
// ---------------------------------------------------------------------------
__global__ __launch_bounds__(32)
void crf_combine_kernel(const unsigned int* __restrict__ chunkBuf,
                        const float* __restrict__ offBuf,
                        const float* __restrict__ trans,
                        float* __restrict__ fwd_out,
                        int nchunks)
{
    const int b    = blockIdx.x;
    const int lane = threadIdx.x;          // output tag i
    const int qq   = (lane & 15) >> 1;     // uint slot for K = lane
    const int kgrp = (lane < 16) ? 0 : 16; // lane-group offset for K = lane

    float fv = (lane == START_TAG) ? 0.0f : -10000.0f;

    for (int c = 0; c < nchunks; ++c) {
        float m = fv;
#pragma unroll
        for (int s = 16; s > 0; s >>= 1) m = fmaxf(m, __shfl_xor(m, s, 32));
        const float ev = nexp(fv - m);
        const unsigned int* base = chunkBuf + (size_t)(b * nchunks + c) * 512;
        float acc = 0.0f;
#pragma unroll
        for (int k = 0; k < 32; ++k) {
            const unsigned int* tb = base + ((k >= 16) ? 256 : 0);
            const unsigned int u = tb[((k & 15) + kgrp) * 8 + qq];
            const unsigned short us = (lane & 1) ? (unsigned short)(u >> 16)
                                                 : (unsigned short)(u & 0xffffu);
            const float pv = (float)__builtin_bit_cast(_Float16, us); // expP[lane, k]
            acc += pv * __shfl(ev, k, 32);
        }
        fv = offBuf[b * nchunks + c] + m + nlog(acc);
    }
    // terminal: logZ = logsumexp_i( fv[i] + trans[STOP][i] )
    fv += trans[STOP_TAG * TAGS + lane];
    float m = fv;
#pragma unroll
    for (int s = 16; s > 0; s >>= 1) m = fmaxf(m, __shfl_xor(m, s, 32));
    float e = nexp(fv - m);
#pragma unroll
    for (int s = 16; s > 0; s >>= 1) e += __shfl_xor(e, s, 32);
    if (lane == 0) fwd_out[b] = m + nlog(e);
}

// ---------------------------------------------------------------------------
// Fallback monolithic forward kernel (used only if d_ws is too small for the
// chunked path). One wave per 16 batches; 2 WMMAs per timestep.
// ---------------------------------------------------------------------------
__global__ __launch_bounds__(32)
void crf_forward_kernel(const float* __restrict__ feats,
                        const float* __restrict__ trans,
                        float* __restrict__ fwd_out)
{
    __shared__ __align__(16) float cbuf[16 * CSTRIDE];
    __shared__ __align__(16) float mbuf[16];

    const int  lane = threadIdx.x;
    const bool lo   = lane < 16;
    const int  col  = lane & 15;
    const int  row  = col;
    const int  Mbase = lo ? 0 : 8;
    const long batch = (long)blockIdx.x * 16 + row;
    const float* featRow = feats + (size_t)batch * SEQ * TAGS;

    v16h b0, b1;
#pragma unroll
    for (int e = 0; e < 16; ++e) {
        const int j = e + (lo ? 0 : 16);
        b0[e] = (_Float16)nexp(trans[(col)      * TAGS + j]);
        b1[e] = (_Float16)nexp(trans[(col + 16) * TAGS + j]);
    }
    const v8f zc = {};
    {
        v16h a = {};
        if (!lo) a[14] = (_Float16)1.0f;   // K = 30 = START_TAG
        v8f d0 = __builtin_amdgcn_wmma_f32_16x16x32_f16(false, a, false, b0, (short)0, zc, false, false);
        v8f d1 = __builtin_amdgcn_wmma_f32_16x16x32_f16(false, a, false, b1, (short)0, zc, false, false);
#pragma unroll
        for (int r = 0; r < 8; ++r) {
            cbuf[(Mbase + r) * CSTRIDE + col]      = nlog(d0[r]);
            cbuf[(Mbase + r) * CSTRIDE + col + 16] = nlog(d1[r]);
        }
    }
    for (int t = 1; t < SEQ; ++t) {
        float x[32];
        const float* fr = featRow + (size_t)(t - 1) * TAGS;
        __builtin_prefetch(fr + 16 * TAGS, 0, 1);
#pragma unroll
        for (int q = 0; q < 8; ++q) {
            v4f c4 = *(const v4f*)(&cbuf[row * CSTRIDE + q * 4]);
            v4f f4 = *(const v4f*)(fr + q * 4);
            x[q*4+0] = c4.x + f4.x;  x[q*4+1] = c4.y + f4.y;
            x[q*4+2] = c4.z + f4.z;  x[q*4+3] = c4.w + f4.w;
        }
        float m = x[0];
#pragma unroll
        for (int j = 1; j < 32; ++j) m = fmaxf(m, x[j]);
        if (lo) mbuf[row] = m;

        v16h a;
#pragma unroll
        for (int e = 0; e < 16; ++e) {
            const int k0 = e;
            const float xa = lo ? x[k0] : x[k0 + 16];
            a[e] = (_Float16)nexp(xa - m);
        }
        v8f d0 = __builtin_amdgcn_wmma_f32_16x16x32_f16(false, a, false, b0, (short)0, zc, false, false);
        v8f d1 = __builtin_amdgcn_wmma_f32_16x16x32_f16(false, a, false, b1, (short)0, zc, false, false);

        float mv[16];
#pragma unroll
        for (int q = 0; q < 4; ++q) {
            v4f m4 = *(const v4f*)(&mbuf[q * 4]);
            mv[q*4+0] = m4.x; mv[q*4+1] = m4.y; mv[q*4+2] = m4.z; mv[q*4+3] = m4.w;
        }
#pragma unroll
        for (int r = 0; r < 8; ++r) {
            const float ms = lo ? mv[r] : mv[r + 8];
            cbuf[(Mbase + r) * CSTRIDE + col]      = ms + nlog(d0[r]);
            cbuf[(Mbase + r) * CSTRIDE + col + 16] = ms + nlog(d1[r]);
        }
    }
    {
        float y[32];
        const float* fr = featRow + (size_t)(SEQ - 1) * TAGS;
#pragma unroll
        for (int q = 0; q < 8; ++q) {
            v4f c4 = *(const v4f*)(&cbuf[row * CSTRIDE + q * 4]);
            v4f f4 = *(const v4f*)(fr + q * 4);
            v4f t4 = *(const v4f*)(trans + STOP_TAG * TAGS + q * 4);
            y[q*4+0] = c4.x + f4.x + t4.x;  y[q*4+1] = c4.y + f4.y + t4.y;
            y[q*4+2] = c4.z + f4.z + t4.z;  y[q*4+3] = c4.w + f4.w + t4.w;
        }
        float m = y[0];
#pragma unroll
        for (int j = 1; j < 32; ++j) m = fmaxf(m, y[j]);
        float s = 0.0f;
#pragma unroll
        for (int j = 0; j < 32; ++j) s += nexp(y[j] - m);
        if (lo) fwd_out[batch] = m + nlog(s);
    }
}

// ---------------------------------------------------------------------------
// Gold-path score + final reduction (deterministic tree reductions).
// ---------------------------------------------------------------------------
__global__ __launch_bounds__(256)
void crf_gold_kernel(const float* __restrict__ feats,
                     const float* __restrict__ trans,
                     const int* __restrict__ tags,
                     float* __restrict__ gold_out)
{
    __shared__ float red[256];
    const int b   = blockIdx.x;
    const int tid = threadIdx.x;
    const int*   tb = tags  + (size_t)b * SEQ;
    const float* fb = feats + (size_t)b * SEQ * TAGS;

    float acc = 0.0f;
    for (int s = tid; s < SEQ; s += 256) {
        const int tg   = tb[s];
        const int prev = (s == 0) ? START_TAG : tb[s - 1];
        acc += trans[tg * TAGS + prev] + fb[(size_t)s * TAGS + tg];
    }
    red[tid] = acc;
    __syncthreads();
#pragma unroll
    for (int w = 128; w > 0; w >>= 1) {
        if (tid < w) red[tid] += red[tid + w];
        __syncthreads();
    }
    if (tid == 0) gold_out[b] = red[0] + trans[STOP_TAG * TAGS + tb[SEQ - 1]];
}

__global__ __launch_bounds__(512)
void crf_final_kernel(const float* __restrict__ fwd,
                      const float* __restrict__ gold,
                      float* __restrict__ out)
{
    __shared__ float red[512];
    const int tid = threadIdx.x;
    red[tid] = fwd[tid] - gold[tid];
    __syncthreads();
#pragma unroll
    for (int w = 256; w > 0; w >>= 1) {
        if (tid < w) red[tid] += red[tid + w];
        __syncthreads();
    }
    if (tid == 0) out[0] = red[0];
}

extern "C" void kernel_launch(void* const* d_in, const int* in_sizes, int n_in,
                              void* d_out, int out_size, void* d_ws, size_t ws_size,
                              hipStream_t stream)
{
    const float* feats = (const float*)d_in[0];   // (512, 2048, 32) f32
    const float* trans = (const float*)d_in[1];   // (32, 32) f32
    const int*   tags  = (const int*)d_in[2];     // (512, 2048) i32

    float* wsF = (float*)d_ws;                    // [512] forward scores
    float* wsG = wsF + BATCH;                     // [512] gold scores

    // Pick the largest chunk count whose scratch footprint fits ws_size.
    int nchunks = 0;
    for (int cand = MAXCHUNKS; cand >= 1; cand >>= 1) {
        const size_t need = (size_t)(2 * BATCH + BATCH * cand) * 4
                          + (size_t)BATCH * cand * 512 * 4;
        if (ws_size >= need) { nchunks = cand; break; }
    }

    if (nchunks > 0) {
        float* offBuf = wsG + BATCH;                                     // [BATCH*nchunks]
        unsigned int* chunkBuf = (unsigned int*)(offBuf + BATCH * nchunks);
        crf_chunk_kernel<<<dim3(nchunks, BATCH), 32, 0, stream>>>(
            feats, trans, chunkBuf, offBuf, SEQ / nchunks);
        crf_combine_kernel<<<BATCH, 32, 0, stream>>>(chunkBuf, offBuf, trans, wsF, nchunks);
    } else {
        crf_forward_kernel<<<BATCH / 16, 32, 0, stream>>>(feats, trans, wsF);
    }
    crf_gold_kernel<<<BATCH, 256, 0, stream>>>(feats, trans, tags, wsG);
    crf_final_kernel<<<1, BATCH, 0, stream>>>(wsF, wsG, (float*)d_out);
}
